// LMFreezeConcatModel_39505109188787
// MI455X (gfx1250) — compile-verified, matrix operand order checked by reading, not run on hardware
//
#include <hip/hip_runtime.h>

// ---------------------------------------------------------------------------
// Types for WMMA
// ---------------------------------------------------------------------------
typedef __attribute__((ext_vector_type(16))) __bf16          v16bf;
typedef __attribute__((ext_vector_type(16))) unsigned short  v16u;
typedef __attribute__((ext_vector_type(8)))  float           v8f;

__device__ __forceinline__ unsigned short f32_to_bf16_rne(float f) {
    unsigned int u = __float_as_uint(f);
    u += 0x7FFFu + ((u >> 16) & 1u);      // round-to-nearest-even
    return (unsigned short)(u >> 16);
}

// Fragment element index for a given k within a 32-wide chunk (16-bit A/B layout):
//   forward:  kk = half*8 + ((e>>1)&3)*2 + (e>>3)*16 + (e&1),  half = lane>>4
//   inverse:  e  = ((kk>>4)<<3) + (((kk>>1)&3)<<1) + (kk&1),   halfbit = (kk>>3)&1
// Consecutive (even,odd) k pairs map to consecutive (even,odd) e pairs -> one b32 store.

// ---------------------------------------------------------------------------
// Generic GEMM: C[N,128] = A[N,K] (f32, row-major) @ B[K,128] (f32, row-major)
//               (+ optional bias[128])
// f32 -> bf16 on the staging path, f32 accumulation via v_wmma_f32_16x16x32_bf16.
// Tiles are stored in LDS in the *fragment register layout*, so each lane loads
// a whole 16-element bf16 fragment as one contiguous 32B LDS read (2x b128).
// One block = 256 threads = 8 wave32 waves; wave w owns rows [16w,16w+16).
// ---------------------------------------------------------------------------
__global__ void __launch_bounds__(256)
gemm_bf16_wmma(const float* __restrict__ A, const float* __restrict__ B,
               const float* __restrict__ bias, float* __restrict__ C,
               int N, int K) {
    const int tid  = threadIdx.x;
    const int wave = tid >> 5;          // 0..7 (wave32)
    const int lane = tid & 31;
    const int row0 = blockIdx.x * 128;

    // Fragment-layout tiles: [fragment][lane] of 16 bf16 (32B) each.
    __shared__ v16u AsF[8][32];         // 8 row-fragments (16 rows each) x 32 lanes
    __shared__ v16u BsF[8][32];         // 8 col-fragments (16 cols each) x 32 lanes

    v8f acc[8] = {};                    // 8 n-tiles of 16x16 f32

    for (int k0 = 0; k0 < K; k0 += 32) {
        // ---- stage A tile (128 rows x 32 k) directly into fragment layout ----
        // p = (row r)*16 + kpair ; consecutive lanes -> consecutive k pairs (coalesced f2)
        for (int p = tid; p < 128 * 16; p += 256) {
            const int r  = p >> 4;
            const int kk = (p & 15) << 1;
            const int row = row0 + r;
            float2 v;
            if (row < N) v = *(const float2*)&A[(size_t)row * K + (k0 + kk)];
            else         { v.x = 0.0f; v.y = 0.0f; }
            const unsigned int packed =
                (unsigned int)f32_to_bf16_rne(v.x) |
                ((unsigned int)f32_to_bf16_rne(v.y) << 16);
            const int w  = r >> 4;
            const int ln = (r & 15) + (((kk >> 3) & 1) << 4);
            const int e  = ((kk >> 4) << 3) + (((kk >> 1) & 3) << 1);
            ((unsigned int*)&AsF[w][ln])[e >> 1] = packed;
        }
        // ---- stage B tile (32 k x 128 cols) into fragment layout ----
        // p = kpair*128 + c ; consecutive lanes -> consecutive columns (coalesced)
        for (int p = tid; p < 16 * 128; p += 256) {
            const int kp = p >> 7;
            const int c  = p & 127;
            const int kk = kp << 1;
            const float v0 = B[(size_t)(k0 + kk)     * 128 + c];
            const float v1 = B[(size_t)(k0 + kk + 1) * 128 + c];
            const unsigned int packed =
                (unsigned int)f32_to_bf16_rne(v0) |
                ((unsigned int)f32_to_bf16_rne(v1) << 16);
            const int nt = c >> 4;
            const int ln = (c & 15) + (((kk >> 3) & 1) << 4);
            const int e  = ((kk >> 4) << 3) + (((kk >> 1) & 3) << 1);
            ((unsigned int*)&BsF[nt][ln])[e >> 1] = packed;
        }
        // speculative prefetch of next A chunk (global_prefetch_b8)
        if (k0 + 32 < K) {
            const int r = tid & 127;
            const int row = row0 + r;
            if (row < N) __builtin_prefetch(&A[(size_t)row * K + (k0 + 32)], 0, 1);
        }
        __syncthreads();

        // ---- fragments: one 32B contiguous LDS read per matrix per lane ----
        const v16bf af = __builtin_bit_cast(v16bf, AsF[wave][lane]);
#pragma unroll
        for (int nt = 0; nt < 8; ++nt) {
            const v16bf bf = __builtin_bit_cast(v16bf, BsF[nt][lane]);
            acc[nt] = __builtin_amdgcn_wmma_f32_16x16x32_bf16(
                false, af, false, bf, (short)0, acc[nt], false, false);
        }
        __syncthreads();
    }

    // ---- epilogue: C/D 16x16 f32 layout (lane 0-15: M=r, lane 16-31: M=8+r) ----
#pragma unroll
    for (int nt = 0; nt < 8; ++nt) {
        const int col = (nt << 4) + (lane & 15);
        const float bv = bias ? bias[col] : 0.0f;
#pragma unroll
        for (int r = 0; r < 8; ++r) {
            const int row = row0 + (wave << 4) + r + ((lane >> 4) << 3);
            if (row < N) C[(size_t)row * 128 + col] = acc[nt][r] + bv;
        }
    }
}

// ---------------------------------------------------------------------------
// Degree / normalization kernels
// ---------------------------------------------------------------------------
__global__ void zero_f32(float* p, int n) {
    int t = blockIdx.x * blockDim.x + threadIdx.x;
    if (t < n) p[t] = 0.0f;
}

__global__ void count_deg(const int* __restrict__ ei, float* __restrict__ deg, int E) {
    int t = blockIdx.x * blockDim.x + threadIdx.x;
    if (t < E) atomicAdd(&deg[ei[E + t]], 1.0f);   // dst = row 1
}

__global__ void make_dinv(float* __restrict__ d, int n) {
    int t = blockIdx.x * blockDim.x + threadIdx.x;
    if (t < n) d[t] = rsqrtf(fmaxf(d[t] + 1.0f, 1.0f));  // +1 self loop
}

// acc[i,:] = dinv[i]^2 * h[i,:]   (self-loop term; also zero-initializes acc)
__global__ void init_acc(const float* __restrict__ h, const float* __restrict__ dinv,
                         float* __restrict__ acc, int N) {
    int t = blockIdx.x * blockDim.x + threadIdx.x;   // N*32 threads, 4 floats each
    if (t >= N * 32) return;
    const int node = t >> 5;
    const int c = (t & 31) << 2;
    float s = dinv[node]; s *= s;
    const float4 hv = *(const float4*)&h[(size_t)node * 128 + c];
    float4 o; o.x = hv.x * s; o.y = hv.y * s; o.z = hv.z * s; o.w = hv.w * s;
    *(float4*)&acc[(size_t)node * 128 + c] = o;
}

// one wave per edge, 4 channels per lane; f32 atomics land in L2 (acc fits in 192MB L2)
__global__ void edge_scatter(const int* __restrict__ ei, const float* __restrict__ dinv,
                             const float* __restrict__ h, float* __restrict__ acc, int E) {
    int t = blockIdx.x * blockDim.x + threadIdx.x;
    const int e = t >> 5;
    if (e >= E) return;
    const int c = (t & 31) << 2;
    const int s = ei[e];          // src (messages flow src -> dst)
    const int d = ei[E + e];      // dst
    const float nrm = dinv[s] * dinv[d];
    const float4 hv = *(const float4*)&h[(size_t)s * 128 + c];
    float* ap = &acc[(size_t)d * 128 + c];
    atomicAdd(ap + 0, hv.x * nrm);
    atomicAdd(ap + 1, hv.y * nrm);
    atomicAdd(ap + 2, hv.z * nrm);
    atomicAdd(ap + 3, hv.w * nrm);
}

__global__ void finalize_relu(const float* __restrict__ acc, const float* __restrict__ b,
                              float* __restrict__ out, int N) {
    int t = blockIdx.x * blockDim.x + threadIdx.x;   // N*128
    if (t >= N * 128) return;
    const float v = acc[t] + b[t & 127];
    out[t] = v > 0.0f ? v : 0.0f;
}

// logits[n,c] = b[c] + sum_k lm[n,k]*W[k,c] + sum_k out[n,k]*W[128+k,c]
__global__ void classifier(const float* __restrict__ lm, const float* __restrict__ out,
                           const float* __restrict__ W, const float* __restrict__ b,
                           float* __restrict__ logits, int N) {
    int t = blockIdx.x * blockDim.x + threadIdx.x;
    if (t >= N * 8) return;
    const int n = t >> 3, c = t & 7;
    float s = b[c];
    const float* lmr = &lm[(size_t)n * 128];
    const float* otr = &out[(size_t)n * 128];
#pragma unroll 4
    for (int k = 0; k < 128; ++k) s += lmr[k] * W[k * 8 + c];
#pragma unroll 4
    for (int k = 0; k < 128; ++k) s += otr[k] * W[(128 + k) * 8 + c];
    logits[t] = s;
}

// ---------------------------------------------------------------------------
// Host launch
// ---------------------------------------------------------------------------
extern "C" void kernel_launch(void* const* d_in, const int* in_sizes, int n_in,
                              void* d_out, int out_size, void* d_ws, size_t ws_size,
                              hipStream_t stream) {
    const float* X     = (const float*)d_in[0];   // [N,1024]
    const int*   ei    = (const int*)  d_in[1];   // [2,E]
    const float* W_lm  = (const float*)d_in[2];   // [1024,128]
    const float* b_lm  = (const float*)d_in[3];   // [128]
    const float* W_gcn = (const float*)d_in[4];   // [3,128,128]
    const float* b_gcn = (const float*)d_in[5];   // [3,128]
    const float* W_cls = (const float*)d_in[6];   // [256,8]
    const float* b_cls = (const float*)d_in[7];   // [8]

    const int D = 1024, H = 128;
    const int N = in_sizes[0] / D;
    const int E = in_sizes[1] / 2;

    float* ws   = (float*)d_ws;
    float* lm   = ws;                        // N*H
    float* h    = lm  + (size_t)N * H;       // N*H
    float* acc  = h   + (size_t)N * H;       // N*H
    float* out  = acc + (size_t)N * H;       // N*H
    float* dinv = out + (size_t)N * H;       // N

    const int T = 256;

    // degrees -> dinv
    zero_f32   <<<(N + T - 1) / T, T, 0, stream>>>(dinv, N);
    count_deg  <<<(E + T - 1) / T, T, 0, stream>>>(ei, dinv, E);
    make_dinv  <<<(N + T - 1) / T, T, 0, stream>>>(dinv, N);

    // lm = X @ W_lm + b_lm   (HBM-bound: 410MB of X at 23.3TB/s)
    gemm_bf16_wmma<<<(N + 127) / 128, T, 0, stream>>>(X, W_lm, b_lm, lm, N, D);

    const float* cur = lm;
    for (int l = 0; l < 3; ++l) {
        gemm_bf16_wmma<<<(N + 127) / 128, T, 0, stream>>>(
            cur, W_gcn + (size_t)l * H * H, nullptr, h, N, H);
        init_acc     <<<((size_t)N * 32 + T - 1) / T, T, 0, stream>>>(h, dinv, acc, N);
        edge_scatter <<<((size_t)E * 32 + T - 1) / T, T, 0, stream>>>(ei, dinv, h, acc, E);
        finalize_relu<<<((size_t)N * 128 + T - 1) / T, T, 0, stream>>>(
            acc, b_gcn + (size_t)l * H, out, N);
        cur = out;
    }

    classifier<<<((size_t)N * 8 + T - 1) / T, T, 0, stream>>>(
        lm, out, W_cls, b_cls, (float*)d_out, N);
}
